// BiologicallyInformedBaseline_82188494176334
// MI455X (gfx1250) — matrix operand order repaired and checked by reading, not compiled, not run on hardware
//
#include <hip/hip_runtime.h>

#define NN 8192
#define NE 262144
#define IN_DIM 256
#define HIDD 256
#define OUTD 64
#define PD 64
#define NH 4
#define DH 16

typedef __attribute__((ext_vector_type(16))) __bf16 v16bf;
typedef __attribute__((ext_vector_type(8)))  float  v8f;
typedef __attribute__((ext_vector_type(8)))  unsigned int v8u;

__device__ __forceinline__ unsigned int f2bf1(float f) {
  unsigned int u = __float_as_uint(f);
  u += 0x7fffu + ((u >> 16) & 1u);   // round-to-nearest-even to bf16
  return u >> 16;
}
__device__ __forceinline__ unsigned int packbf2(float lo, float hi) {
  return f2bf1(lo) | (f2bf1(hi) << 16);
}
__device__ __forceinline__ v8f wmma_bf16(v8u a, v8u b, v8f c) {
  return __builtin_amdgcn_wmma_f32_16x16x32_bf16(
      false, __builtin_bit_cast(v16bf, a),
      false, __builtin_bit_cast(v16bf, b),
      (short)0, c, false, false);
}

// ---------------------------------------------------------------------------
// bf16-WMMA GEMM: C[M,N] = A[M,K] @ B + bias (opt), relu (opt).
// transB==0: B is [K,N] row-major.  transB==1: B is [N,K] row-major (W.T GEMM)
// Block = 128 threads (4 waves); block tile 64x64; each wave owns a 32x32
// subtile = 4 accumulators / 2 A-frags / 2 B-frags -> 4 WMMAs per K-step of 32
// with 2x fragment reuse. Requires M,N multiples of 64 and K multiple of 32.
// ---------------------------------------------------------------------------
__global__ void __launch_bounds__(128)
gemm_bf16(const float* __restrict__ A, const float* __restrict__ B,
          const float* __restrict__ bias, float* __restrict__ C,
          int M, int N, int K, int ldc, int transB, int relu) {
  __shared__ unsigned int sA[64][16];   // [m][k/2] packed bf16 pairs (4 KB)
  __shared__ unsigned int sB[64][16];   // [n][k/2] packed bf16 pairs (4 KB)
  const int tid  = threadIdx.x;
  const int wave = tid >> 5, lane = tid & 31;
  const int half = lane >> 4, lm = lane & 15;
  const int m0 = blockIdx.y * 64, n0 = blockIdx.x * 64;
  const int sm = (wave >> 1) * 32, sn = (wave & 1) * 32;
  v8f acc00 = {}, acc01 = {}, acc10 = {}, acc11 = {};
  const int nk = K >> 5;
  for (int kb = 0; kb < nk; ++kb) {
    const int k0 = kb << 5;
    __syncthreads();
#pragma unroll
    for (int i = 0; i < 8; ++i) {
      int idx = tid + i * 128;          // 0..1023
      int r = idx >> 4, kpi = idx & 15; // r: tile row (A) / tile col (B)
      sA[r][kpi] = packbf2(A[(m0 + r) * K + k0 + 2 * kpi],
                           A[(m0 + r) * K + k0 + 2 * kpi + 1]);
      float b0, b1;
      if (transB) {
        b0 = B[(n0 + r) * K + k0 + 2 * kpi];
        b1 = B[(n0 + r) * K + k0 + 2 * kpi + 1];
      } else {
        b0 = B[(k0 + 2 * kpi) * N + n0 + r];
        b1 = B[(k0 + 2 * kpi + 1) * N + n0 + r];
      }
      sB[r][kpi] = packbf2(b0, b1);
    }
    if (kb + 1 < nk) {                  // exercise global_prefetch_b8
      __builtin_prefetch(&A[(m0 + (tid >> 1)) * K + k0 + 32], 0, 3);
    }
    __syncthreads();
    v8u a0, a1, b0, b1;
#pragma unroll
    for (int v = 0; v < 8; ++v) {
      int ka = 8 * (v >> 2) + 4 * half + (v & 3);   // A-frag packed index
      int kbp = 8 * (v >> 2) + 4 * half + (v & 3);  // B-frag packed index
      a0[v] = sA[sm + lm][ka];
      a1[v] = sA[sm + 16 + lm][ka];
      b0[v] = sB[sn + lm][kbp];
      b1[v] = sB[sn + 16 + lm][kbp];
    }
    acc00 = wmma_bf16(a0, b0, acc00);
    acc01 = wmma_bf16(a0, b1, acc01);
    acc10 = wmma_bf16(a1, b0, acc10);
    acc11 = wmma_bf16(a1, b1, acc11);
  }
#pragma unroll
  for (int ti = 0; ti < 2; ++ti) {
#pragma unroll
    for (int tj = 0; tj < 2; ++tj) {
      v8f acc = (ti == 0) ? ((tj == 0) ? acc00 : acc01)
                          : ((tj == 0) ? acc10 : acc11);
#pragma unroll
      for (int r = 0; r < 8; ++r) {
        int row = m0 + sm + ti * 16 + r + 8 * half;
        int col = n0 + sn + tj * 16 + lm;
        float v = acc[r] + (bias ? bias[col] : 0.0f);
        if (relu) v = fmaxf(v, 0.0f);
        C[row * ldc + col] = v;
      }
    }
  }
}

// ---------------------------------------------------------------------------
// Flash attention over qkv [NN, 3*PD], heads of dh=16, softmax scale 1/4.
// One wave handles one (query-tile-of-16, head). Key chunks of 64:
// 4 S-WMMAs (Q-frag reused 4x, dh=16 zero-padded to K=32), one online-softmax
// pass over the 4 score fragments (intra-half xor shuffles), P tile (16x64)
// re-swizzled through per-wave LDS, then O += P@V as 2 WMMAs.
// ---------------------------------------------------------------------------
__global__ void __launch_bounds__(128)
flash_attn(const float* __restrict__ qkv, float* __restrict__ out) {
  __shared__ unsigned int pbuf[4][16][32];  // per-wave P tile 16x64 bf16 (8 KB)
  const int lane = threadIdx.x & 31;
  const int wv = threadIdx.x >> 5;
  const int wid = blockIdx.x * 4 + wv;
  const int head = wid & (NH - 1);
  const int qt = wid >> 2;
  const int half = lane >> 4, lm = lane & 15;
  const int hc = head * DH;
  const float* qp = qkv;
  const float* kp = qkv + PD;
  const float* vp = qkv + 2 * PD;
  const int ldq = 3 * PD;

  // Q fragment (A layout): only K dims 0..15 valid (dh=16), rest zero.
  v8u qa = {};
  const int qrow = qt * 16 + lm;
#pragma unroll
  for (int v = 0; v < 4; ++v) {
    int kd = 8 * half + 2 * v;
    qa[v] = packbf2(qp[qrow * ldq + hc + kd], qp[qrow * ldq + hc + kd + 1]);
  }
  float rm[8], rl[8];
#pragma unroll
  for (int r = 0; r < 8; ++r) { rm[r] = -1e30f; rl[r] = 0.0f; }
  v8f o = {};

  for (int cb = 0; cb < NN / 64; ++cb) {
    const int kb0 = cb * 64;
    // Scores for 4 key sub-blocks of 16. K^T fragments (B layout): elements
    // 8..15 cover K dims 16..31 of the WMMA -> zero (dh=16).
    v8f s[4];
#pragma unroll
    for (int t = 0; t < 4; ++t) {
      v8u kt = {};
      {
#pragma unroll
        for (int v = 0; v < 4; ++v) {
          int kr = (kb0 + t * 16 + lm) * ldq + hc + 8 * half + 2 * v;
          kt[v] = packbf2(kp[kr], kp[kr + 1]);
        }
      }
      v8f zero = {};
      s[t] = wmma_bf16(qa, kt, zero);
    }

    unsigned short* ps = (unsigned short*)&pbuf[wv][0][0];
#pragma unroll
    for (int r = 0; r < 8; ++r) {
      float a0 = s[0][r] * 0.25f;
      float a1 = s[1][r] * 0.25f;
      float a2 = s[2][r] * 0.25f;
      float a3 = s[3][r] * 0.25f;
      float mx = fmaxf(fmaxf(a0, a1), fmaxf(a2, a3));
#pragma unroll
      for (int d = 1; d < 16; d <<= 1) mx = fmaxf(mx, __shfl_xor(mx, d, 32));
      float nm = fmaxf(rm[r], mx);
      float p0 = __expf(a0 - nm);
      float p1 = __expf(a1 - nm);
      float p2 = __expf(a2 - nm);
      float p3 = __expf(a3 - nm);
      float rs = (p0 + p1) + (p2 + p3);
#pragma unroll
      for (int d = 1; d < 16; d <<= 1) rs += __shfl_xor(rs, d, 32);
      float corr = __expf(rm[r] - nm);
      rl[r] = rl[r] * corr + rs;
      rm[r] = nm;
      o[r] *= corr;
      int prow = r + 8 * half;                 // C-layout row of this VGPR
      ps[prow * 64 + lm]      = (unsigned short)f2bf1(p0);
      ps[prow * 64 + 16 + lm] = (unsigned short)f2bf1(p1);
      ps[prow * 64 + 32 + lm] = (unsigned short)f2bf1(p2);
      ps[prow * 64 + 48 + lm] = (unsigned short)f2bf1(p3);
    }
    // P as two A fragments (keys 0..31 and 32..63 of the chunk) from LDS.
    v8u pa0, pa1;
#pragma unroll
    for (int v = 0; v < 8; ++v) {
      int ka = 8 * (v >> 2) + 4 * half + (v & 3);
      pa0[v] = pbuf[wv][lm][ka];
      pa1[v] = pbuf[wv][lm][16 + ka];
    }
    // V fragments (B layout over 32 keys each).
    v8u vb0, vb1;
#pragma unroll
    for (int v = 0; v < 8; ++v) {
      int vr = kb0 + 2 * (8 * (v >> 2) + 4 * half + (v & 3));
      vb0[v] = packbf2(vp[vr * ldq + hc + lm], vp[(vr + 1) * ldq + hc + lm]);
      vb1[v] = packbf2(vp[(vr + 32) * ldq + hc + lm], vp[(vr + 33) * ldq + hc + lm]);
    }
    o = wmma_bf16(pa0, vb0, o);
    o = wmma_bf16(pa1, vb1, o);
  }
#pragma unroll
  for (int r = 0; r < 8; ++r) {
    int row = qt * 16 + r + 8 * half;
    out[row * PD + hc + lm] = o[r] / rl[r];
  }
}

// ----------------------------- GCN helpers ---------------------------------
__global__ void copy_x_hcat(const float* __restrict__ x, float* __restrict__ hcat) {
  int i = blockIdx.x * blockDim.x + threadIdx.x;
  if (i >= NN * IN_DIM) return;
  int n = i / IN_DIM, c = i % IN_DIM;
  hcat[n * (IN_DIM + PD) + c] = x[i];
}
__global__ void deg_init(float* __restrict__ deg) {
  int i = blockIdx.x * blockDim.x + threadIdx.x;
  if (i < NN) deg[i] = 1.0f;                      // self loop
}
__global__ void deg_count(const int* __restrict__ ei, float* __restrict__ deg) {
  int e = blockIdx.x * blockDim.x + threadIdx.x;
  if (e < NE) atomicAdd(&deg[ei[NE + e]], 1.0f);  // dst counts
}
__global__ void make_dinv(float* __restrict__ deg) {
  int i = blockIdx.x * blockDim.x + threadIdx.x;
  if (i < NN) deg[i] = rsqrtf(deg[i]);            // in place: deg -> dinv
}
__global__ void self_init(const float* __restrict__ hw, const float* __restrict__ dinv,
                          float* __restrict__ agg, int N) {
  int i = blockIdx.x * blockDim.x + threadIdx.x;
  if (i >= NN * N) return;
  float di = dinv[i / N];
  agg[i] = hw[i] * di * di;
}
__global__ void edge_scatter(const int* __restrict__ ei, const float* __restrict__ hw,
                             const float* __restrict__ dinv, float* __restrict__ agg, int N) {
  int i = blockIdx.x * blockDim.x + threadIdx.x;
  if (i >= NE * N) return;
  int e = i / N, c = i % N;
  int s = ei[e], d = ei[NE + e];
  atomicAdd(&agg[d * N + c], hw[s * N + c] * dinv[s] * dinv[d]);
}
__global__ void bias_act(const float* __restrict__ agg, const float* __restrict__ b,
                         float* __restrict__ out, int N, int relu) {
  int i = blockIdx.x * blockDim.x + threadIdx.x;
  if (i >= NN * N) return;
  float v = agg[i] + b[i % N];
  if (relu) v = fmaxf(v, 0.0f);
  out[i] = v;
}

extern "C" void kernel_launch(void* const* d_in, const int* in_sizes, int n_in,
                              void* d_out, int out_size, void* d_ws, size_t ws_size,
                              hipStream_t stream) {
  (void)in_sizes; (void)n_in; (void)out_size; (void)ws_size;
  const float* x   = (const float*)d_in[0];
  const int*   ei  = (const int*)d_in[1];
  const float* pew = (const float*)d_in[2];
  const float* peb = (const float*)d_in[3];
  const float* ipw = (const float*)d_in[4];
  const float* ipb = (const float*)d_in[5];
  const float* opw = (const float*)d_in[6];
  const float* opb = (const float*)d_in[7];
  const float* w1  = (const float*)d_in[8];
  const float* b1  = (const float*)d_in[9];
  const float* w2  = (const float*)d_in[10];
  const float* b2  = (const float*)d_in[11];
  const float* w3  = (const float*)d_in[12];
  const float* b3  = (const float*)d_in[13];
  float* outp = (float*)d_out;

  float* ws   = (float*)d_ws;
  float* pf   = ws;                         // 8192*64 ; later reused as attn out
  float* qkv  = pf + NN * PD;               // 8192*192
  float* attn = pf;                         // alias (pf dead after qkv GEMM)
  float* hcat = qkv;                        // 8192*320 (qkv dead after attention)
  float* hw   = hcat + NN * (IN_DIM + PD);  // 8192*256
  float* aggA = hw + NN * HIDD;             // 8192*256
  float* aggB = aggA + NN * HIDD;           // 8192*256
  float* dinv = aggB + NN * HIDD;           // 8192

  dim3 blk(128);
  // 1) pf = x @ pe_w + pe_b
  gemm_bf16<<<dim3(PD / 64, NN / 64), blk, 0, stream>>>(
      x, pew, peb, pf, NN, PD, IN_DIM, PD, 0, 0);
  // 2) qkv = pf @ in_proj_w.T + in_proj_b
  gemm_bf16<<<dim3(3 * PD / 64, NN / 64), blk, 0, stream>>>(
      pf, ipw, ipb, qkv, NN, 3 * PD, PD, 3 * PD, 1, 0);
  // 3) multi-head self-attention (flash)
  flash_attn<<<dim3((NN / 16) * NH / 4), blk, 0, stream>>>(qkv, attn);
  // 4) pa = attn @ out_proj_w.T + out_proj_b  -> hcat columns [256,320)
  gemm_bf16<<<dim3(PD / 64, NN / 64), blk, 0, stream>>>(
      attn, opw, opb, hcat + IN_DIM, NN, PD, PD, IN_DIM + PD, 1, 0);
  // 5) hcat columns [0,256) = x
  copy_x_hcat<<<(NN * IN_DIM + 255) / 256, 256, 0, stream>>>(x, hcat);
  // 6) symmetric GCN normalization (once; shared by all three layers)
  deg_init<<<(NN + 255) / 256, 256, 0, stream>>>(dinv);
  deg_count<<<(NE + 255) / 256, 256, 0, stream>>>(ei, dinv);
  make_dinv<<<(NN + 255) / 256, 256, 0, stream>>>(dinv);
  // ---- GCN layer 1: relu(conv(hcat, w1) + b1) -> aggA
  gemm_bf16<<<dim3(HIDD / 64, NN / 64), blk, 0, stream>>>(
      hcat, w1, nullptr, hw, NN, HIDD, IN_DIM + PD, HIDD, 0, 0);
  self_init<<<(NN * HIDD + 255) / 256, 256, 0, stream>>>(hw, dinv, aggA, HIDD);
  edge_scatter<<<(NE * HIDD + 255) / 256, 256, 0, stream>>>(ei, hw, dinv, aggA, HIDD);
  bias_act<<<(NN * HIDD + 255) / 256, 256, 0, stream>>>(aggA, b1, aggA, HIDD, 1);
  // ---- GCN layer 2: relu(conv(aggA, w2) + b2) -> aggB
  gemm_bf16<<<dim3(HIDD / 64, NN / 64), blk, 0, stream>>>(
      aggA, w2, nullptr, hw, NN, HIDD, HIDD, HIDD, 0, 0);
  self_init<<<(NN * HIDD + 255) / 256, 256, 0, stream>>>(hw, dinv, aggB, HIDD);
  edge_scatter<<<(NE * HIDD + 255) / 256, 256, 0, stream>>>(ei, hw, dinv, aggB, HIDD);
  bias_act<<<(NN * HIDD + 255) / 256, 256, 0, stream>>>(aggB, b2, aggB, HIDD, 1);
  // ---- GCN layer 3: conv(aggB, w3) + b3 -> d_out
  gemm_bf16<<<dim3(OUTD / 64, NN / 64), blk, 0, stream>>>(
      aggB, w3, nullptr, hw, NN, OUTD, HIDD, OUTD, 0, 0);
  self_init<<<(NN * OUTD + 255) / 256, 256, 0, stream>>>(hw, dinv, aggA, OUTD);
  edge_scatter<<<(NE * OUTD + 255) / 256, 256, 0, stream>>>(ei, hw, dinv, aggA, OUTD);
  bias_act<<<(NN * OUTD + 255) / 256, 256, 0, stream>>>(aggA, b3, outp, OUTD, 0);
}
